// TransMIL_19078244729088
// MI455X (gfx1250) — compile-verified, matrix operand order checked by reading, not run on hardware
//
#include <hip/hip_runtime.h>
#include <hip/hip_bf16.h>

// ---------------------------------------------------------------------------
// TransMIL (Nystrom attention) for MI455X / gfx1250, wave32 + WMMA.
//   - All big GEMMs: v_wmma_f32_16x16x32_f16 (f16 operands, f32 accum),
//     all operands consumed in coalesced W-form (C = A @ W^T); small RHS
//     matrices are produced transposed by GEMM epilogues so no strided loads.
//   - Pinv Newton-Schulz GEMMs: v_wmma_f32_16x16x4_f32 (full f32).
// Shapes: B=2, NP=4096 tokens (incl cls), D=512, H=8, DH=64, M=256, l=16.
// ---------------------------------------------------------------------------

typedef __attribute__((ext_vector_type(16))) _Float16 v16h;
typedef __attribute__((ext_vector_type(8)))  float    v8f;
typedef __attribute__((ext_vector_type(2)))  float    v2f;

#if defined(__has_builtin)
#if __has_builtin(__builtin_amdgcn_wmma_f32_16x16x4_f32)
#define HAVE_WMMA_F32F32 1
#endif
#endif

// ----------------------------- WMMA helpers --------------------------------

__device__ __forceinline__ v8f wmma16(v16h a, v16h b, v8f c) {
  return __builtin_amdgcn_wmma_f32_16x16x32_f16(false, a, false, b, (short)0, c,
                                                false, false);
}

// A-frag (16x32, f16) from row-major [*, ld] at (row0, k0). Rows clamped.
// Per-lane K map (ISA 7.12.2): lane<16 -> M=lane, Ks {0..7,16..23};
// lane>=16 -> same M, K += 8.
__device__ __forceinline__ v16h load_frag_rm16(const _Float16* __restrict__ P,
                                               int ld, int row0, int k0,
                                               int rmax) {
  int lane = threadIdx.x & 31;
  int r = row0 + (lane & 15);
  if (r > rmax) r = rmax;
  int hi = lane >> 4;
  const _Float16* base = P + (long)r * ld + k0 + 8 * hi;
  v16h f;
#pragma unroll
  for (int p = 0; p < 8; ++p) {
    int kb = 2 * (p & 3) + 16 * (p >> 2);
    f[2 * p]     = base[kb];
    f[2 * p + 1] = base[kb + 1];
  }
  return f;
}

// 32x32 output tile, C = A @ W^T; A:[M,K] row-major, W:[N,K] row-major.
__device__ __forceinline__ void tile32x32(const _Float16* A, int lda, int rmax,
                                          const _Float16* W, int ldw, int nmax,
                                          int m0, int n0, int K,
                                          v8f (&acc)[2][2]) {
  for (int k0 = 0; k0 < K; k0 += 32) {
    v16h a0 = load_frag_rm16(A, lda, m0, k0, rmax);
    v16h a1 = load_frag_rm16(A, lda, m0 + 16, k0, rmax);
    v16h b0 = load_frag_rm16(W, ldw, n0, k0, nmax);
    v16h b1 = load_frag_rm16(W, ldw, n0 + 16, k0, nmax);
    acc[0][0] = wmma16(a0, b0, acc[0][0]);
    acc[0][1] = wmma16(a0, b1, acc[0][1]);
    acc[1][0] = wmma16(a1, b0, acc[1][0]);
    acc[1][1] = wmma16(a1, b1, acc[1][1]);
  }
}

__device__ __forceinline__ void zero_acc(v8f (&acc)[2][2]) {
#pragma unroll
  for (int i = 0; i < 2; ++i)
#pragma unroll
    for (int j = 0; j < 2; ++j)
#pragma unroll
      for (int v = 0; v < 8; ++v) acc[i][j][v] = 0.0f;
}

// ----------------------------- GEMM kernels --------------------------------

// fc1: H[b, tok+1, :] = relu(x @ W^T + bias); x16:[8190,1024], W:[512,1024]
__global__ void k_gemm_fc1(const _Float16* __restrict__ A,
                           const _Float16* __restrict__ W,
                           const float* __restrict__ bias,
                           float* __restrict__ H) {
  const int M = 2 * 4095, K = 1024;
  int m0 = blockIdx.y * 32, n0 = blockIdx.x * 32;
  v8f acc[2][2];
  zero_acc(acc);
  tile32x32(A, K, M - 1, W, K, 511, m0, n0, K, acc);
  int lane = threadIdx.x & 31, hi = lane >> 4, c0 = lane & 15;
#pragma unroll
  for (int mi = 0; mi < 2; ++mi)
#pragma unroll
    for (int ni = 0; ni < 2; ++ni)
#pragma unroll
      for (int v = 0; v < 8; ++v) {
        int row = m0 + 16 * mi + v + 8 * hi;
        int col = n0 + 16 * ni + c0;
        if (row < M) {
          float val = acc[mi][ni][v] + bias[col];
          val = val > 0.0f ? val : 0.0f;
          int b = row / 4095, t = row % 4095;
          H[((long)b * 4096 + t + 1) * 512 + col] = val;
        }
      }
}

// qkv: split heads; q scaled by DH^-0.5 = 0.125. A:[8192,512], W:[1536,512].
// v is written transposed (vt16: [bh,64,4096]) for the a3@v GEMM, plus f32
// copy in [bh,4096,64] for the depthwise conv.
__global__ void k_gemm_qkv(const _Float16* __restrict__ A,
                           const _Float16* __restrict__ W,
                           _Float16* __restrict__ q16,
                           _Float16* __restrict__ k16,
                           _Float16* __restrict__ vt16,
                           float* __restrict__ v32) {
  int m0 = blockIdx.y * 32, n0 = blockIdx.x * 32;
  v8f acc[2][2];
  zero_acc(acc);
  tile32x32(A, 512, 8191, W, 512, 1535, m0, n0, 512, acc);
  int lane = threadIdx.x & 31, hi = lane >> 4, c0 = lane & 15;
#pragma unroll
  for (int mi = 0; mi < 2; ++mi)
#pragma unroll
    for (int ni = 0; ni < 2; ++ni)
#pragma unroll
      for (int v = 0; v < 8; ++v) {
        int row = m0 + 16 * mi + v + 8 * hi;   // global token
        int col = n0 + 16 * ni + c0;           // 0..1535
        int b = row >> 12, n = row & 4095;
        int which = col >> 9, head = (col >> 6) & 7, dh = col & 63;
        long bh = (long)(b * 8 + head);
        float val = acc[mi][ni][v];
        if (which == 0)
          q16[(bh * 4096 + n) * 64 + dh] = (_Float16)(val * 0.125f);
        else if (which == 1)
          k16[(bh * 4096 + n) * 64 + dh] = (_Float16)val;
        else {
          vt16[(bh * 64 + dh) * 4096 + n] = (_Float16)val;
          v32[(bh * 4096 + n) * 64 + dh] = val;
        }
      }
}

// out projection with residual add: H += A @ W^T + bias
__global__ void k_gemm_outproj(const _Float16* __restrict__ A,
                               const _Float16* __restrict__ W,
                               const float* __restrict__ bias,
                               float* __restrict__ H) {
  int m0 = blockIdx.y * 32, n0 = blockIdx.x * 32;
  v8f acc[2][2];
  zero_acc(acc);
  tile32x32(A, 512, 8191, W, 512, 511, m0, n0, 512, acc);
  int lane = threadIdx.x & 31, hi = lane >> 4, c0 = lane & 15;
#pragma unroll
  for (int mi = 0; mi < 2; ++mi)
#pragma unroll
    for (int ni = 0; ni < 2; ++ni)
#pragma unroll
      for (int v = 0; v < 8; ++v) {
        int row = m0 + 16 * mi + v + 8 * hi;
        int col = n0 + 16 * ni + c0;
        H[(long)row * 512 + col] += acc[mi][ni][v] + bias[col];
      }
}

// Generic batched C = A @ W^T.
// mode 0: f32 C[row*ldc+col]; mode 1: f16 C[row*ldc+col];
// mode 2: f16 transposed C[col*ldc+row] (produces W-form for next GEMM).
__global__ void k_gemm_bt(const _Float16* __restrict__ A,
                          const _Float16* __restrict__ W,
                          float* __restrict__ Cf, _Float16* __restrict__ Ch,
                          int M, int N, int K, int lda, int ldw, int ldc,
                          long sA, long sW, long sC, int mode) {
  A += (long)blockIdx.z * sA;
  W += (long)blockIdx.z * sW;
  long cbase = (long)blockIdx.z * sC;
  int m0 = blockIdx.y * 32, n0 = blockIdx.x * 32;
  v8f acc[2][2];
  zero_acc(acc);
  tile32x32(A, lda, M - 1, W, ldw, N - 1, m0, n0, K, acc);
  int lane = threadIdx.x & 31, hi = lane >> 4, c0 = lane & 15;
#pragma unroll
  for (int mi = 0; mi < 2; ++mi)
#pragma unroll
    for (int ni = 0; ni < 2; ++ni)
#pragma unroll
      for (int v = 0; v < 8; ++v) {
        int row = m0 + 16 * mi + v + 8 * hi;
        int col = n0 + 16 * ni + c0;
        if (row < M && col < N) {
          float val = acc[mi][ni][v];
          if (mode == 0)      Cf[cbase + (long)row * ldc + col] = val;
          else if (mode == 1) Ch[cbase + (long)row * ldc + col] = (_Float16)val;
          else                Ch[cbase + (long)col * ldc + row] = (_Float16)val;
        }
      }
}

// ---------------- f32 WMMA batched 256x256x256 (pinv iterations) -----------
// C = cE * E + cAcc * (A @ B); all row-major [256,256], batch stride 65536.

__device__ __forceinline__ v2f load_a32(const float* __restrict__ P, int row0,
                                        int k0) {
  int lane = threadIdx.x & 31;
  const float* base = P + (long)(row0 + (lane & 15)) * 256 + k0 + 2 * (lane >> 4);
  v2f f;
  f[0] = base[0];
  f[1] = base[1];
  return f;
}
__device__ __forceinline__ v2f load_b32(const float* __restrict__ P, int k0,
                                        int col0) {
  int lane = threadIdx.x & 31;
  const float* base = P + (long)(k0 + 2 * (lane >> 4)) * 256 + col0 + (lane & 15);
  v2f f;
  f[0] = base[0];
  f[1] = base[256];
  return f;
}

__global__ void k_gemm256(const float* __restrict__ A,
                          const float* __restrict__ B,
                          const float* __restrict__ E, float* __restrict__ C,
                          float cE, float cAcc) {
  long off = (long)blockIdx.z * 65536;
  A += off; B += off; E += off; C += off;
  int m0 = blockIdx.y * 32, n0 = blockIdx.x * 32;
  v8f acc[2][2];
  zero_acc(acc);
#if HAVE_WMMA_F32F32
#pragma unroll 8
  for (int k0 = 0; k0 < 256; k0 += 4) {
    v2f a0 = load_a32(A, m0, k0);
    v2f a1 = load_a32(A, m0 + 16, k0);
    v2f b0 = load_b32(B, k0, n0);
    v2f b1 = load_b32(B, k0, n0 + 16);
    acc[0][0] = __builtin_amdgcn_wmma_f32_16x16x4_f32(false, a0, false, b0,
                                                      (short)0, acc[0][0],
                                                      false, false);
    acc[0][1] = __builtin_amdgcn_wmma_f32_16x16x4_f32(false, a0, false, b1,
                                                      (short)0, acc[0][1],
                                                      false, false);
    acc[1][0] = __builtin_amdgcn_wmma_f32_16x16x4_f32(false, a1, false, b0,
                                                      (short)0, acc[1][0],
                                                      false, false);
    acc[1][1] = __builtin_amdgcn_wmma_f32_16x16x4_f32(false, a1, false, b1,
                                                      (short)0, acc[1][1],
                                                      false, false);
  }
  int lane = threadIdx.x & 31, hi = lane >> 4, c0 = lane & 15;
#pragma unroll
  for (int mi = 0; mi < 2; ++mi)
#pragma unroll
    for (int ni = 0; ni < 2; ++ni)
#pragma unroll
      for (int v = 0; v < 8; ++v) {
        int row = m0 + 16 * mi + v + 8 * hi;
        int col = n0 + 16 * ni + c0;
        long idx = (long)row * 256 + col;
        C[idx] = cE * E[idx] + cAcc * acc[mi][ni][v];
      }
#else
  // scalar fallback (compile-safe path if the f32 WMMA builtin is absent)
  int lane = threadIdx.x & 31, hi = lane >> 4, c0 = lane & 15;
#pragma unroll
  for (int mi = 0; mi < 2; ++mi)
#pragma unroll
    for (int ni = 0; ni < 2; ++ni)
      for (int v = 0; v < 8; ++v) {
        int row = m0 + 16 * mi + v + 8 * hi;
        int col = n0 + 16 * ni + c0;
        float s = 0.0f;
        for (int k = 0; k < 256; ++k)
          s += A[(long)row * 256 + k] * B[(long)k * 256 + col];
        long idx = (long)row * 256 + col;
        C[idx] = cE * E[idx] + cAcc * s;
      }
#endif
}

// ----------------------------- small kernels -------------------------------

__device__ __forceinline__ float block_sum256(float v, float* red) {
  int t = threadIdx.x;
  red[t] = v;
  __syncthreads();
  for (int s = 128; s > 0; s >>= 1) {
    if (t < s) red[t] += red[t + s];
    __syncthreads();
  }
  float r = red[0];
  __syncthreads();
  return r;
}

__global__ void k_cvt(const float* __restrict__ s, _Float16* __restrict__ d,
                      long n) {
  long i = (long)blockIdx.x * 256 + threadIdx.x;
  if (i < n) d[i] = (_Float16)s[i];
}

__global__ void k_cls(const float* __restrict__ cls, float* __restrict__ H) {
  int t = blockIdx.x * 256 + threadIdx.x;
  if (t < 1024) {
    int b = t >> 9, d = t & 511;
    H[(long)b * 4096 * 512 + d] = cls[d];
  }
}

__global__ void k_layernorm(const float* __restrict__ X,
                            const float* __restrict__ w,
                            const float* __restrict__ b,
                            _Float16* __restrict__ Y) {
  __shared__ float red[256];
  long row = blockIdx.x;
  const float* x = X + row * 512;
  int t = threadIdx.x;
  float v0 = x[t], v1 = x[t + 256];
  float mean = block_sum256(v0 + v1, red) * (1.0f / 512.0f);
  float d0 = v0 - mean, d1 = v1 - mean;
  float var = block_sum256(d0 * d0 + d1 * d1, red) * (1.0f / 512.0f);
  float rstd = rsqrtf(var + 1e-5f);
  Y[row * 512 + t]       = (_Float16)(d0 * rstd * w[t] + b[t]);
  Y[row * 512 + t + 256] = (_Float16)(d1 * rstd * w[t + 256] + b[t + 256]);
}

// landmark means: L[bh, m, d] = mean over 16 consecutive tokens of X[bh, n, d]
__global__ void k_landmark(const _Float16* __restrict__ X,
                           _Float16* __restrict__ L) {
  int idx = blockIdx.x * 256 + threadIdx.x;  // 16*256*64 = 262144
  if (idx >= 16 * 256 * 64) return;
  int d = idx & 63, mi = (idx >> 6) & 255, bh = idx >> 14;
  const _Float16* p = X + ((long)bh * 4096 + mi * 16) * 64 + d;
  float s = 0.0f;
#pragma unroll
  for (int t = 0; t < 16; ++t) s += (float)p[t * 64];
  L[idx] = (_Float16)(s * (1.0f / 16.0f));
}

// row softmax; output f16 (oh) or f32 (of)
__global__ void k_softmax(const float* __restrict__ S, _Float16* __restrict__ oh,
                          float* __restrict__ of, int C) {
  __shared__ float red[256];
  long row = blockIdx.x;
  const float* x = S + row * (long)C;
  int t = threadIdx.x;
  float mx = -3.4e38f;
  for (int j = t; j < C; j += 256) mx = fmaxf(mx, x[j]);
  red[t] = mx;
  __syncthreads();
  for (int s = 128; s > 0; s >>= 1) {
    if (t < s) red[t] = fmaxf(red[t], red[t + s]);
    __syncthreads();
  }
  mx = red[0];
  __syncthreads();
  float sum = 0.0f;
  for (int j = t; j < C; j += 256) sum += __expf(x[j] - mx);
  float inv = 1.0f / block_sum256(sum, red);
  for (int j = t; j < C; j += 256) {
    float e = __expf(x[j] - mx) * inv;
    if (oh) oh[row * (long)C + j] = (_Float16)e;
    else    of[row * (long)C + j] = e;
  }
}

// abs row/col sums for pinv init; X:[16][256][256]
__global__ void k_abssum(const float* __restrict__ X, float* __restrict__ out,
                         int transposed) {
  __shared__ float red[256];
  int bh = blockIdx.x >> 8, r = blockIdx.x & 255;
  const float* base = X + (long)bh * 65536;
  int t = threadIdx.x;
  float s = transposed ? fabsf(base[(long)t * 256 + r])
                       : fabsf(base[(long)r * 256 + t]);
  float tot = block_sum256(s, red);
  if (t == 0) out[blockIdx.x] = tot;
}

__global__ void k_maxred(const float* __restrict__ rs,
                         const float* __restrict__ cs, float* __restrict__ scal) {
  __shared__ float r1[256];
  __shared__ float r2[256];
  int t = threadIdx.x;
  float m1 = -1.0f, m2 = -1.0f;
  for (int i = t; i < 4096; i += 256) {
    m1 = fmaxf(m1, rs[i]);
    m2 = fmaxf(m2, cs[i]);
  }
  r1[t] = m1;
  r2[t] = m2;
  __syncthreads();
  for (int s = 128; s > 0; s >>= 1) {
    if (t < s) {
      r1[t] = fmaxf(r1[t], r1[t + s]);
      r2[t] = fmaxf(r2[t], r2[t + s]);
    }
    __syncthreads();
  }
  if (t == 0) {
    scal[0] = r1[0];
    scal[1] = r2[0];
  }
}

// z0 = x^T / (col_max * row_max)
__global__ void k_pinv_init(const float* __restrict__ X,
                            const float* __restrict__ scal,
                            float* __restrict__ Z) {
  int idx = blockIdx.x * 256 + threadIdx.x;  // 16*65536
  if (idx >= 16 * 65536) return;
  int j = idx & 255, i = (idx >> 8) & 255, bh = idx >> 16;
  float inv = 1.0f / (scal[0] * scal[1]);
  Z[idx] = X[(long)bh * 65536 + (long)j * 256 + i] * inv;
}

// depthwise residual conv (kernel 33 over sequence) + head merge to [b,n,512]
__global__ void k_conv_merge(const float* __restrict__ AO,
                             const float* __restrict__ V,
                             const float* __restrict__ RW,
                             _Float16* __restrict__ Y) {
  int idx = blockIdx.x * 256 + threadIdx.x;  // 16*4096*64
  if (idx >= 16 * 4096 * 64) return;
  int d = idx & 63, n = (idx >> 6) & 4095, bh = idx >> 18;
  int h = bh & 7, b = bh >> 3;
  float acc = AO[idx];
  const float* vb = V + (long)bh * 4096 * 64 + d;
  const float* rw = RW + h * 33;
#pragma unroll
  for (int t = 0; t < 33; ++t) {
    int sn = n + t - 16;
    if (sn >= 0 && sn < 4096) acc += vb[(long)sn * 64] * rw[t];
  }
  Y[((long)(b * 4096 + n)) * 512 + h * 64 + d] = (_Float16)acc;
}

// final LN on cls row + fc2
__global__ void k_head(const float* __restrict__ H, const float* __restrict__ w,
                       const float* __restrict__ bn,
                       const float* __restrict__ fc2w,
                       const float* __restrict__ fc2b, float* __restrict__ out) {
  __shared__ float xn[512];
  __shared__ float red[256];
  int b = blockIdx.x, t = threadIdx.x;
  const float* x = H + (long)b * 4096 * 512;
  float v0 = x[t], v1 = x[t + 256];
  float mean = block_sum256(v0 + v1, red) * (1.0f / 512.0f);
  float d0 = v0 - mean, d1 = v1 - mean;
  float var = block_sum256(d0 * d0 + d1 * d1, red) * (1.0f / 512.0f);
  float rstd = rsqrtf(var + 1e-5f);
  xn[t] = d0 * rstd * w[t] + bn[t];
  xn[t + 256] = d1 * rstd * w[t + 256] + bn[t + 256];
  __syncthreads();
  for (int o = t; o < 1000; o += 256) {
    float s = fc2b[o];
    const float* wr = fc2w + (long)o * 512;
    for (int k = 0; k < 512; ++k) s += xn[k] * wr[k];
    out[b * 1000 + o] = s;
  }
}

// ------------------------------- driver ------------------------------------

extern "C" void kernel_launch(void* const* d_in, const int* in_sizes, int n_in,
                              void* d_out, int out_size, void* d_ws,
                              size_t ws_size, hipStream_t stream) {
  (void)in_sizes; (void)n_in; (void)out_size; (void)ws_size;

  const float* X     = (const float*)d_in[0];
  const float* FC1W  = (const float*)d_in[1];
  const float* FC1B  = (const float*)d_in[2];
  const float* CLS   = (const float*)d_in[3];
  const float* LNW[2]  = {(const float*)d_in[4],  (const float*)d_in[10]};
  const float* LNB[2]  = {(const float*)d_in[5],  (const float*)d_in[11]};
  const float* QKVW[2] = {(const float*)d_in[6],  (const float*)d_in[12]};
  const float* OUTW[2] = {(const float*)d_in[7],  (const float*)d_in[13]};
  const float* OUTB[2] = {(const float*)d_in[8],  (const float*)d_in[14]};
  const float* RESW[2] = {(const float*)d_in[9],  (const float*)d_in[15]};
  const float* NW    = (const float*)d_in[16];
  const float* NB    = (const float*)d_in[17];
  const float* FC2W  = (const float*)d_in[18];
  const float* FC2B  = (const float*)d_in[19];
  float* OUT = (float*)d_out;

  char* base = (char*)d_ws;
  size_t off = 0;
  auto carve = [&](size_t bytes) -> void* {
    void* r = base + off;
    off += (bytes + 255) & ~(size_t)255;
    return r;
  };

  const long BHND = 16L * 4096 * 64;   // 4,194,304
  _Float16* x16  = (_Float16*)carve(8190L * 1024 * 2);
  _Float16* w16  = (_Float16*)carve(1536L * 512 * 2);
  float*    h    = (float*)   carve(8192L * 512 * 4);
  _Float16* ln16 = (_Float16*)carve(8192L * 512 * 2);
  _Float16* q16  = (_Float16*)carve(BHND * 2);
  _Float16* k16  = (_Float16*)carve(BHND * 2);
  _Float16* vt16 = (_Float16*)carve(BHND * 2);        // [bh, 64, 4096]
  float*    v32  = (float*)   carve(BHND * 4);        // [bh, 4096, 64]
  _Float16* ql16 = (_Float16*)carve(16L * 256 * 64 * 2);
  _Float16* kl16 = (_Float16*)carve(16L * 256 * 64 * 2);
  float*    sim  = (float*)   carve(16L * 4096 * 256 * 4);
  _Float16* a1   = (_Float16*)carve(16L * 4096 * 256 * 2);
  _Float16* a3   = (_Float16*)carve(16L * 256 * 4096 * 2);
  float*    x2   = (float*)   carve(16L * 65536 * 4);
  float*    z0   = (float*)   carve(16L * 65536 * 4);
  float*    z1   = (float*)   carve(16L * 65536 * 4);
  float*    xz   = (float*)   carve(16L * 65536 * 4);
  float*    tb   = (float*)   carve(16L * 65536 * 4);
  float*    tc   = (float*)   carve(16L * 65536 * 4);
  _Float16* z16  = (_Float16*)carve(16L * 65536 * 2);
  _Float16* t3t  = (_Float16*)carve(16L * 16384 * 2); // [bh, 64, 256]
  _Float16* p2t  = (_Float16*)carve(16L * 16384 * 2); // [bh, 64, 256]
  float*    aout = (float*)   carve(16L * 4096 * 64 * 4);
  _Float16* ao16 = (_Float16*)carve(8192L * 512 * 2);
  float*    rsum = (float*)   carve(4096 * 4);
  float*    csum = (float*)   carve(4096 * 4);
  float*    scal = (float*)   carve(2 * 4);

  // --- fc1 + cls ---
  {
    long n = 8190L * 1024;
    k_cvt<<<dim3((unsigned)((n + 255) / 256)), 256, 0, stream>>>(X, x16, n);
  }
  k_cvt<<<dim3((512 * 1024 + 255) / 256), 256, 0, stream>>>(FC1W, w16,
                                                            512L * 1024);
  k_cls<<<dim3(4), 256, 0, stream>>>(CLS, h);
  k_gemm_fc1<<<dim3(16, 256), 32, 0, stream>>>(x16, w16, FC1B, h);

  for (int L = 0; L < 2; ++L) {
    // layernorm -> f16
    k_layernorm<<<dim3(8192), 256, 0, stream>>>(h, LNW[L], LNB[L], ln16);
    // qkv projection with head split (v stored transposed + f32 copy)
    k_cvt<<<dim3((1536 * 512 + 255) / 256), 256, 0, stream>>>(QKVW[L], w16,
                                                              1536L * 512);
    k_gemm_qkv<<<dim3(48, 256), 32, 0, stream>>>(ln16, w16, q16, k16, vt16,
                                                 v32);
    // landmarks (mean over groups of 16)
    k_landmark<<<dim3(1024), 256, 0, stream>>>(q16, ql16);
    k_landmark<<<dim3(1024), 256, 0, stream>>>(k16, kl16);
    // sim1 = q @ k_l^T -> softmax -> a1 (f16)
    k_gemm_bt<<<dim3(8, 128, 16), 32, 0, stream>>>(
        q16, kl16, sim, nullptr, 4096, 256, 64, 64, 64, 256, 262144L, 16384L,
        1048576L, 0);
    k_softmax<<<dim3(65536), 256, 0, stream>>>(sim, a1, nullptr, 256);
    // sim2 = q_l @ k_l^T -> softmax -> x2 (f32)
    k_gemm_bt<<<dim3(8, 8, 16), 32, 0, stream>>>(
        ql16, kl16, xz, nullptr, 256, 256, 64, 64, 64, 256, 16384L, 16384L,
        65536L, 0);
    k_softmax<<<dim3(4096), 256, 0, stream>>>(xz, nullptr, x2, 256);
    // sim3 = q_l @ k^T -> softmax -> a3 (f16)
    k_gemm_bt<<<dim3(128, 8, 16), 32, 0, stream>>>(
        ql16, k16, sim, nullptr, 256, 4096, 64, 64, 64, 4096, 16384L, 262144L,
        1048576L, 0);
    k_softmax<<<dim3(4096), 256, 0, stream>>>(sim, a3, nullptr, 4096);
    // pinv init: z0 = x^T / (max rowsum * max colsum)
    k_abssum<<<dim3(4096), 256, 0, stream>>>(x2, rsum, 0);
    k_abssum<<<dim3(4096), 256, 0, stream>>>(x2, csum, 1);
    k_maxred<<<dim3(1), 256, 0, stream>>>(rsum, csum, scal);
    k_pinv_init<<<dim3(4096), 256, 0, stream>>>(x2, scal, z0);
    // 6 Newton-Schulz iterations, 4 fused GEMMs each:
    //   xz = x@z ; tb = 7xz - xz@xz ; tc = 15xz - xz@tb ; z' = 3.25z - 0.25 z@tc
    float* zc = z0;
    float* zn = z1;
    for (int it = 0; it < 6; ++it) {
      k_gemm256<<<dim3(8, 8, 16), 32, 0, stream>>>(x2, zc, x2, xz, 0.0f, 1.0f);
      k_gemm256<<<dim3(8, 8, 16), 32, 0, stream>>>(xz, xz, xz, tb, 7.0f, -1.0f);
      k_gemm256<<<dim3(8, 8, 16), 32, 0, stream>>>(xz, tb, xz, tc, 15.0f, -1.0f);
      k_gemm256<<<dim3(8, 8, 16), 32, 0, stream>>>(zc, tc, zc, zn, 3.25f,
                                                   -0.25f);
      float* t = zc; zc = zn; zn = t;
    }
    // t3^T = (a3 @ v)^T : A=a3 [256,4096], W=vt16 [64,4096] -> t3t [64,256] f16
    k_gemm_bt<<<dim3(2, 8, 16), 32, 0, stream>>>(
        a3, vt16, nullptr, t3t, 256, 64, 4096, 4096, 4096, 256, 1048576L,
        262144L, 16384L, 2);
    // p2^T = (z @ t3)^T : A=z16 [256,256], W=t3t [64,256] -> p2t [64,256] f16
    k_cvt<<<dim3((16 * 65536 + 255) / 256), 256, 0, stream>>>(zc, z16,
                                                              16L * 65536);
    k_gemm_bt<<<dim3(2, 8, 16), 32, 0, stream>>>(
        z16, t3t, nullptr, p2t, 256, 64, 256, 256, 256, 256, 65536L, 16384L,
        16384L, 2);
    // attn out = a1 @ p2 : A=a1 [4096,256], W=p2t [64,256] -> aout f32
    k_gemm_bt<<<dim3(2, 128, 16), 32, 0, stream>>>(
        a1, p2t, aout, nullptr, 4096, 64, 256, 256, 256, 64, 1048576L, 16384L,
        262144L, 0);
    // + depthwise conv residual over sequence, merge heads -> [b,n,512] f16
    k_conv_merge<<<dim3(16384), 256, 0, stream>>>(aout, v32, RESW[L], ao16);
    // out projection with residual add into h
    k_cvt<<<dim3((512 * 512 + 255) / 256), 256, 0, stream>>>(OUTW[L], w16,
                                                             512L * 512);
    k_gemm_outproj<<<dim3(16, 256), 32, 0, stream>>>(ao16, w16, OUTB[L], h);
  }

  // final LN on cls token + fc2
  k_head<<<dim3(2), 256, 0, stream>>>(h, NW, NB, FC2W, FC2B, OUT);
}
